// SimplifiedMultiHeadSelfAttention_79628693667935
// MI455X (gfx1250) — matrix operand order repaired and checked by reading, not compiled, not run on hardware
//
#include <hip/hip_runtime.h>

// ---------------------------------------------------------------------------
// SimplifiedMultiHeadSelfAttention for MI455X (gfx1250, wave32, WMMA f16)
// B=2 S=2048 D=1024 H=16 HD=64
// ---------------------------------------------------------------------------

typedef __attribute__((ext_vector_type(16))) _Float16 v16h;
typedef __attribute__((ext_vector_type(8)))  _Float16 v8h;
typedef __attribute__((ext_vector_type(4)))  _Float16 v4h;
typedef __attribute__((ext_vector_type(8)))  float    v8f;

#define B_   2
#define S_   2048
#define D_   1024
#define H_   16
#define HD_  64
#define M_   (B_ * S_)        // 4096 rows
#define SCALE_ 0.125f         // HD^-0.5
#define EPS_ 1e-5f

// ---- WMMA helpers ---------------------------------------------------------

__device__ __forceinline__ v8f wmma_f16(v16h a, v16h b, v8f c) {
  // D = A(16x32 f16) * B(32x16 f16) + C(16x16 f32)
  return __builtin_amdgcn_wmma_f32_16x16x32_f16(false, a, false, b,
                                                (short)0, c, false, false);
}

// A-fragment: lane holds row (lane%16); 16 contiguous K halves:
// lanes 0-15: K[0..15]; lanes 16-31: K[16..31]  (caller adds +16)
__device__ __forceinline__ v16h ld_a(const _Float16* p) {
  v8h lo = *(const v8h*)(p);
  v8h hi = *(const v8h*)(p + 8);
  return __builtin_shufflevector(lo, hi, 0,1,2,3,4,5,6,7,8,9,10,11,12,13,14,15);
}

// B-fragment (loaded from B^T rows): lane holds col n = lane%16; 16 contiguous
// K halves (lanes 0-15: K[0..15]; lanes 16-31: K[16..31] — caller adds +16)
__device__ __forceinline__ v16h ld_b(const _Float16* p) {
  v8h lo = *(const v8h*)(p);
  v8h hi = *(const v8h*)(p + 8);
  return __builtin_shufflevector(lo, hi, 0,1,2,3,4,5,6,7,8,9,10,11,12,13,14,15);
}

// ---- kernel 0: f32 -> f16 convert (4 elems / thread) ----------------------

__global__ __launch_bounds__(256) void cvt_f16(const float* __restrict__ src,
                                               _Float16* __restrict__ dst) {
  const int i = (blockIdx.x * 256 + threadIdx.x) * 4;
  float4 f = *(const float4*)(src + i);
  v4h h;
  h[0] = (_Float16)f.x; h[1] = (_Float16)f.y;
  h[2] = (_Float16)f.z; h[3] = (_Float16)f.w;
  *(v4h*)(dst + i) = h;
}

// ---- kernel 1: weight transpose+convert: WT[n][k] = (f16) W[k][n] ---------

__global__ __launch_bounds__(256) void wt_cvt(const float* __restrict__ W,
                                              _Float16* __restrict__ WT) {
  __shared__ float t[16][17];
  const int tx = threadIdx.x & 15, ty = threadIdx.x >> 4;
  const int nt = blockIdx.x, kt = blockIdx.y;   // 64 x 64 tiles of 16
  t[ty][tx] = W[(size_t)(kt * 16 + ty) * D_ + nt * 16 + tx];
  __syncthreads();
  WT[(size_t)(nt * 16 + ty) * D_ + kt * 16 + tx] = (_Float16)t[tx][ty];
}

// ---- kernel 2: fused QKV projection ---------------------------------------
// One wave computes a 16x64 tile of  out = X @ W + b.
// which=0 -> Q16 [B,H,S,HD]; which=1 -> K16 [B,H,S,HD]; which=2 -> VT16 [B,H,HD,S]

__global__ __launch_bounds__(32) void qkv_gemm(
    const _Float16* __restrict__ X16, const _Float16* __restrict__ WT,
    const float* __restrict__ bq, const float* __restrict__ bk,
    const float* __restrict__ bv,
    _Float16* __restrict__ Q16, _Float16* __restrict__ K16,
    _Float16* __restrict__ VT16) {
  const int lane = threadIdx.x, col = lane & 15, grp = lane >> 4;
  const int mt = blockIdx.x, nt = blockIdx.y, which = blockIdx.z;

  const _Float16* Xb = X16 + (size_t)which * ((size_t)M_ * D_);
  const _Float16* Wb = WT + (size_t)which * ((size_t)D_ * D_);
  const float* bias = (which == 0) ? bq : ((which == 1) ? bk : bv);

  const _Float16* arow = Xb + (size_t)(mt * 16 + col) * D_;
  const _Float16* br0 = Wb + (size_t)(nt * 64 + 0  + col) * D_;
  const _Float16* br1 = Wb + (size_t)(nt * 64 + 16 + col) * D_;
  const _Float16* br2 = Wb + (size_t)(nt * 64 + 32 + col) * D_;
  const _Float16* br3 = Wb + (size_t)(nt * 64 + 48 + col) * D_;

  v8f c0 = {0,0,0,0,0,0,0,0}, c1 = c0, c2 = c0, c3 = c0;
  for (int kc = 0; kc < D_; kc += 32) {
    v16h a  = ld_a(arow + kc + grp * 16);
    v16h b0 = ld_b(br0 + kc + grp * 16);
    v16h b1 = ld_b(br1 + kc + grp * 16);
    v16h b2 = ld_b(br2 + kc + grp * 16);
    v16h b3 = ld_b(br3 + kc + grp * 16);
    c0 = wmma_f16(a, b0, c0);
    c1 = wmma_f16(a, b1, c1);
    c2 = wmma_f16(a, b2, c2);
    c3 = wmma_f16(a, b3, c3);
  }

  const int b = mt >> 7;               // 128 m-tiles per batch
  const int sbase = (mt & 127) * 16;
  v8f cc[4] = {c0, c1, c2, c3};
#pragma unroll
  for (int f = 0; f < 4; ++f) {
    const int n = nt * 64 + f * 16 + col;
    const int h = n >> 6, d = n & 63;
    const float bb = bias[n];
    if (which == 2) {
      // V transposed: VT16[((b*H+h)*HD + d)*S + s], s contiguous per lane
      _Float16* dst = VT16 + ((size_t)((b * H_ + h) * HD_ + d)) * S_ +
                      sbase + 8 * grp;
      v8h o;
#pragma unroll
      for (int v = 0; v < 8; ++v) o[v] = (_Float16)(cc[f][v] + bb);
      *(v8h*)dst = o;
    } else {
      _Float16* dst = (which == 0) ? Q16 : K16;
#pragma unroll
      for (int v = 0; v < 8; ++v) {
        const int s = sbase + v + 8 * grp;
        dst[((size_t)((b * H_ + h) * S_ + s)) * HD_ + d] =
            (_Float16)(cc[f][v] + bb);
      }
    }
  }
}

// ---- kernel 3: flash attention --------------------------------------------
// One wave per (b,h, 16-query tile). Scores computed transposed
// (S^T = K_chunk @ Q^T) so softmax stats reduce along lanes cheaply and the
// exp(P) tile stores to LDS row-major with one b128 per fragment.

__global__ __launch_bounds__(32) void attn_kernel(
    const _Float16* __restrict__ Q16, const _Float16* __restrict__ K16,
    const _Float16* __restrict__ VT16, _Float16* __restrict__ AO16) {
  const int lane = threadIdx.x, col = lane & 15, grp = lane >> 4;
  const int qt = blockIdx.x;   // 0..127
  const int bh = blockIdx.y;   // 0..31 = b*H + h

  const _Float16* Qp = Q16 + (size_t)bh * S_ * HD_;
  const _Float16* Kp = K16 + (size_t)bh * S_ * HD_;
  const _Float16* Vp = VT16 + (size_t)bh * HD_ * S_;

  __shared__ alignas(16) _Float16 P[16 * 40];   // 16 q rows x 32 keys, pad 40

  // Q as B-fragments (held for the whole key loop)
  const _Float16* qrow = Qp + (size_t)(qt * 16 + col) * HD_;
  const v16h bq0 = ld_b(qrow + grp * 16);        // hd 0..31
  const v16h bq1 = ld_b(qrow + 32 + grp * 16);   // hd 32..63

  float mrun = -3.0e38f, lrun = 0.0f;
  v8f O0 = {0,0,0,0,0,0,0,0}, O1 = O0, O2 = O0, O3 = O0;

  for (int kb = 0; kb < S_; kb += 32) {
    const _Float16* k0 = Kp + (size_t)(kb + col) * HD_;        // keys 0..15
    const _Float16* k1 = Kp + (size_t)(kb + 16 + col) * HD_;   // keys 16..31
    __builtin_prefetch(k0 + 32 * HD_, 0, 3);                   // next chunk

    v8f s0 = {0,0,0,0,0,0,0,0}, s1 = s0;
    {
      v16h a00 = ld_a(k0 + grp * 16);        // hd 0..31
      v16h a01 = ld_a(k0 + 32 + grp * 16);   // hd 32..63
      v16h a10 = ld_a(k1 + grp * 16);
      v16h a11 = ld_a(k1 + 32 + grp * 16);
      s0 = wmma_f16(a00, bq0, s0);  s0 = wmma_f16(a01, bq1, s0);
      s1 = wmma_f16(a10, bq0, s1);  s1 = wmma_f16(a11, bq1, s1);
    }

    // scale + per-query chunk max (q = col; partner lane is col+16)
    float mx = -3.0e38f;
#pragma unroll
    for (int i = 0; i < 8; ++i) {
      s0[i] *= SCALE_; s1[i] *= SCALE_;
      mx = fmaxf(mx, fmaxf(s0[i], s1[i]));
    }
    mx = fmaxf(mx, __shfl_xor(mx, 16, 32));
    const float mnew = fmaxf(mrun, mx);
    const float alpha = __expf(mrun - mnew);

    float rs = 0.0f;
    v8h p0, p1;
#pragma unroll
    for (int i = 0; i < 8; ++i) {
      const float ea = __expf(s0[i] - mnew);
      const float eb = __expf(s1[i] - mnew);
      rs += ea + eb;
      p0[i] = (_Float16)ea; p1[i] = (_Float16)eb;
    }
    rs += __shfl_xor(rs, 16, 32);
    lrun = lrun * alpha + rs;
    mrun = mnew;

    // P row-major: row q=col, keys 8*grp+[0..7] / 16+8*grp+[0..7]
    *(v8h*)&P[col * 40 + grp * 8] = p0;
    *(v8h*)&P[col * 40 + 16 + grp * 8] = p1;

    // rescale running O by alpha (broadcast per q-row)
    float ar[8];
#pragma unroll
    for (int v = 0; v < 8; ++v) ar[v] = __shfl(alpha, v + 8 * grp, 32);
#pragma unroll
    for (int v = 0; v < 8; ++v) {
      O0[v] *= ar[v]; O1[v] *= ar[v]; O2[v] *= ar[v]; O3[v] *= ar[v];
    }

    // O += P(16x32) @ V_chunk(32x64)
    v16h pa = ld_a(&P[col * 40 + grp * 16]);
    v16h bv0 = ld_b(Vp + (size_t)(0  + col) * S_ + kb + grp * 16);
    v16h bv1 = ld_b(Vp + (size_t)(16 + col) * S_ + kb + grp * 16);
    v16h bv2 = ld_b(Vp + (size_t)(32 + col) * S_ + kb + grp * 16);
    v16h bv3 = ld_b(Vp + (size_t)(48 + col) * S_ + kb + grp * 16);
    O0 = wmma_f16(pa, bv0, O0);
    O1 = wmma_f16(pa, bv1, O1);
    O2 = wmma_f16(pa, bv2, O2);
    O3 = wmma_f16(pa, bv3, O3);
  }

  // normalize and write attn output as f16 [B,S,D] rows for the O-projection
  const float linv = 1.0f / lrun;
  float lr[8];
#pragma unroll
  for (int v = 0; v < 8; ++v) lr[v] = __shfl(linv, v + 8 * grp, 32);

  const int b = bh >> 4, h = bh & 15;
  const size_t mbase = (size_t)(b * S_ + qt * 16);
  v8f OO[4] = {O0, O1, O2, O3};
#pragma unroll
  for (int f = 0; f < 4; ++f) {
    const int n = h * HD_ + f * 16 + col;
#pragma unroll
    for (int v = 0; v < 8; ++v) {
      const int r = v + 8 * grp;
      AO16[(mbase + r) * D_ + n] = (_Float16)(OO[f][v] * lr[v]);
    }
  }
}

// ---- kernel 4: output projection + bias + residual (f32) ------------------

__global__ __launch_bounds__(32) void oproj_gemm(
    const _Float16* __restrict__ AO, const _Float16* __restrict__ WTo,
    const float* __restrict__ bo, const float* __restrict__ resid,
    float* __restrict__ Y) {
  const int lane = threadIdx.x, col = lane & 15, grp = lane >> 4;
  const int mt = blockIdx.x, nt = blockIdx.y;

  const _Float16* arow = AO + (size_t)(mt * 16 + col) * D_;
  const _Float16* br0 = WTo + (size_t)(nt * 64 + 0  + col) * D_;
  const _Float16* br1 = WTo + (size_t)(nt * 64 + 16 + col) * D_;
  const _Float16* br2 = WTo + (size_t)(nt * 64 + 32 + col) * D_;
  const _Float16* br3 = WTo + (size_t)(nt * 64 + 48 + col) * D_;

  v8f c0 = {0,0,0,0,0,0,0,0}, c1 = c0, c2 = c0, c3 = c0;
  for (int kc = 0; kc < D_; kc += 32) {
    v16h a  = ld_a(arow + kc + grp * 16);
    v16h b0 = ld_b(br0 + kc + grp * 16);
    v16h b1 = ld_b(br1 + kc + grp * 16);
    v16h b2 = ld_b(br2 + kc + grp * 16);
    v16h b3 = ld_b(br3 + kc + grp * 16);
    c0 = wmma_f16(a, b0, c0);
    c1 = wmma_f16(a, b1, c1);
    c2 = wmma_f16(a, b2, c2);
    c3 = wmma_f16(a, b3, c3);
  }

  v8f cc[4] = {c0, c1, c2, c3};
#pragma unroll
  for (int f = 0; f < 4; ++f) {
    const int n = nt * 64 + f * 16 + col;
    const float bb = bo[n];
#pragma unroll
    for (int v = 0; v < 8; ++v) {
      const size_t m = (size_t)(mt * 16 + v + 8 * grp);
      Y[m * D_ + n] = cc[f][v] + bb + resid[m * D_ + n];
    }
  }
}

// ---- kernel 5: LayerNorm over D=1024 --------------------------------------

__global__ __launch_bounds__(256) void ln_kernel(
    const float* __restrict__ Y, const float* __restrict__ gamma,
    const float* __restrict__ beta, float* __restrict__ out) {
  const int row = blockIdx.x, tid = threadIdx.x;
  const float* y = Y + (size_t)row * D_;
  float vals[4], s = 0.0f, s2 = 0.0f;
#pragma unroll
  for (int i = 0; i < 4; ++i) {
    const float v = y[tid + i * 256];
    vals[i] = v; s += v; s2 += v * v;
  }
  __shared__ float shA[256], shB[256];
  shA[tid] = s; shB[tid] = s2;
  __syncthreads();
  for (int off = 128; off > 0; off >>= 1) {
    if (tid < off) { shA[tid] += shA[tid + off]; shB[tid] += shB[tid + off]; }
    __syncthreads();
  }
  const float mu = shA[0] * (1.0f / D_);
  const float var = shB[0] * (1.0f / D_) - mu * mu;
  const float inv = rsqrtf(var + EPS_);
#pragma unroll
  for (int i = 0; i < 4; ++i) {
    const int c = tid + i * 256;
    out[(size_t)row * D_ + c] = (vals[i] - mu) * inv * gamma[c] + beta[c];
  }
}

// ---- host launcher --------------------------------------------------------

extern "C" void kernel_launch(void* const* d_in, const int* in_sizes, int n_in,
                              void* d_out, int out_size, void* d_ws,
                              size_t ws_size, hipStream_t stream) {
  const float* q     = (const float*)d_in[0];
  const float* k     = (const float*)d_in[1];
  const float* v     = (const float*)d_in[2];
  const float* Wq    = (const float*)d_in[3];
  const float* bq    = (const float*)d_in[4];
  const float* Wk    = (const float*)d_in[5];
  const float* bk    = (const float*)d_in[6];
  const float* Wv    = (const float*)d_in[7];
  const float* bv    = (const float*)d_in[8];
  const float* Wo    = (const float*)d_in[9];
  const float* bo    = (const float*)d_in[10];
  const float* gamma = (const float*)d_in[11];
  const float* beta  = (const float*)d_in[12];
  float* out = (float*)d_out;

  // workspace layout (bytes)
  char* ws = (char*)d_ws;
  _Float16* X16  = (_Float16*)(ws + 0);          // 3 x 4096x1024 f16 (25.2MB)
  _Float16* WT   = (_Float16*)(ws + 25165824);   // 4 x 1024x1024 f16 (8.4MB)
  _Float16* Q16  = (_Float16*)(ws + 33554432);   // [B,H,S,HD]
  _Float16* K16  = (_Float16*)(ws + 41943040);   // [B,H,S,HD]
  _Float16* VT16 = (_Float16*)(ws + 50331648);   // [B,H,HD,S]
  _Float16* AO16 = (_Float16*)(ws + 58720256);   // [B,S,D] f16
  float*    Yf   = (float*)(ws + 67108864);      // [B,S,D] f32

  // f32 -> f16 inputs (4096 blocks * 256 thr * 4 elems = 4,194,304)
  cvt_f16<<<4096, 256, 0, stream>>>(q, X16 + 0 * (size_t)M_ * D_);
  cvt_f16<<<4096, 256, 0, stream>>>(k, X16 + 1 * (size_t)M_ * D_);
  cvt_f16<<<4096, 256, 0, stream>>>(v, X16 + 2 * (size_t)M_ * D_);

  // weights -> transposed f16
  dim3 tg(64, 64);
  wt_cvt<<<tg, 256, 0, stream>>>(Wq, WT + 0 * (size_t)D_ * D_);
  wt_cvt<<<tg, 256, 0, stream>>>(Wk, WT + 1 * (size_t)D_ * D_);
  wt_cvt<<<tg, 256, 0, stream>>>(Wv, WT + 2 * (size_t)D_ * D_);
  wt_cvt<<<tg, 256, 0, stream>>>(Wo, WT + 3 * (size_t)D_ * D_);

  // fused QKV projection: grid (m-tiles, n-tiles, which)
  qkv_gemm<<<dim3(256, 16, 3), 32, 0, stream>>>(X16, WT, bq, bk, bv,
                                                Q16, K16, VT16);

  // flash attention: grid (q-tiles, B*H)
  attn_kernel<<<dim3(128, 32), 32, 0, stream>>>(Q16, K16, VT16, AO16);

  // output projection + residual
  oproj_gemm<<<dim3(256, 16), 32, 0, stream>>>(AO16, WT + 3 * (size_t)D_ * D_,
                                               bo, q, Yf);

  // LayerNorm
  ln_kernel<<<M_, 256, 0, stream>>>(Yf, gamma, beta, out);
}
